// MHLocAttention_79860621902535
// MI455X (gfx1250) — matrix operand order repaired and checked by reading, not compiled, not run on hardware
//
#include <hip/hip_runtime.h>

// ---------------------------------------------------------------------------
// Multi-head location-aware attention, fused for MI455X (gfx1250, wave32).
//
// Pipeline:
//   k0 : dec_part[n,h,d] = dec_prev @ W_dec                 (tiny VALU GEMM)
//   k1 : per (n,h,64-t tile): key = enc@W_key, att = conv(ali)@att_w^T,
//        e[t] = w_w . tanh(key+att+dec)    -> bf16 WMMA with register-resident
//        A fragments and software-pipelined B fragments; only the score
//        vector hits HBM.
//   k2 : masked softmax over T -> ali (output #1), and
//        we[n,h,:] = sum_t ali * enc_pad[n,t,:]   (enc_part never formed,
//        valid because sum_t ali == 1 after softmax)
//   k3a: ctx_head = we @ W_enc(head cols) + b_enc           (tiny VALU GEMM)
//   k3b: ctx = ctx_head @ W_ctx + b_ctx (output #2)         (tiny VALU GEMM)
// ---------------------------------------------------------------------------

#define N_   16
#define T_   1500
#define ENC_ 512
#define DEC_ 512
#define D_   512
#define H_   4
#define C_   128
#define K_   11
#define PAD_ 5

// ---- kernel-1 tiling ----
constexpr int TB       = 64;                     // timesteps per workgroup
constexpr int NBLK     = (T_ + TB - 1) / TB;     // 24 tiles cover T=1500
constexpr int THREADS1 = 128;                    // 4 wave32 per workgroup
constexpr int ENC_P    = ENC_ + 8;               // padded LDS stride (bf16 elems)
constexpr int C_P      = C_ + 8;                 // padded LDS stride (bf16 elems)
constexpr int KC_ENC   = ENC_ / 32;              // 16 enc K-chunks
constexpr int KC_F     = C_ / 32;                // 4 conv-feature K-chunks
constexpr int KC_ALL   = KC_ENC + KC_F;          // 20 chunks total

// LDS carve (bytes, all 16B aligned)
constexpr int OFF_ENC = 0;                                   // [TB][ENC_P] bf16
constexpr int OFF_F   = OFF_ENC + TB * ENC_P * 2;            // [TB][C_P]   bf16
constexpr int OFF_WK  = OFF_F   + TB * C_P * 2;              // [16][ENC_P] bf16
constexpr int OFF_WA  = OFF_WK  + 16 * ENC_P * 2;            // [16][C_P]   bf16
constexpr int OFF_FW  = OFF_WA  + 16 * C_P * 2;              // [C_][12]    f32
constexpr int OFF_ALI = OFF_FW  + C_ * 12 * 4;               // [TB+K_-1]   f32
constexpr int SMEM1   = ((OFF_ALI + (TB + K_ - 1) * 4) + 63) & ~63;

typedef __attribute__((ext_vector_type(16))) __bf16 v16bf;
typedef __attribute__((ext_vector_type(8)))  float  v8f;

union Frag32 {          // 32-byte WMMA operand built from two 16B LDS loads
    uint4 u[2];
    v16bf v;
};

__device__ __forceinline__ unsigned short f2bf(float x) {
    unsigned int u = __float_as_uint(x);
    u += 0x7FFFu + ((u >> 16) & 1u);             // round-to-nearest-even
    return (unsigned short)(u >> 16);
}

// A fragment (16-bit A, 16x32): per-lane two 16B segments 32 elems apart / 2
__device__ __forceinline__ void load_a(const unsigned short* p, Frag32& f) {
    f.u[0] = *(const uint4*)p;          // K +0..7   (this lane's K-half)
    f.u[1] = *(const uint4*)(p + 16);   // K +16..23 (this lane's K-half)
}
// B fragment (16-bit B, 32x16): per-lane two 8-K segments 16 apart of one column
__device__ __forceinline__ void load_b(const unsigned short* p, Frag32& f) {
    f.u[0] = *(const uint4*)p;
    f.u[1] = *(const uint4*)(p + 16);
}

// ---------------------------------------------------------------------------
// k0: dec_part[n, h*D+d] = dec_prev[n,:] @ W_dec[:, h*D+d]
// ---------------------------------------------------------------------------
__global__ __launch_bounds__(256) void k0_dec_proj(
    const float* __restrict__ dec_prev, const float* __restrict__ Wdec,
    float* __restrict__ dec_ws)
{
    const int gid = blockIdx.x * 256 + threadIdx.x;     // N*H*D = 32768
    const int n   = gid >> 11;
    const int col = gid & 2047;
    const float* x = dec_prev + (size_t)n * DEC_;
    float acc = 0.f;
    #pragma unroll 8
    for (int k = 0; k < DEC_; ++k)
        acc += x[k] * Wdec[(size_t)k * (H_ * D_) + col];
    dec_ws[gid] = acc;
}

// ---------------------------------------------------------------------------
// k1: fused score kernel.  One workgroup = (n, h, 64-timestep tile).
//     Wave w owns t-rows [16w, 16w+16).  The 20 A fragments (16 enc chunks +
//     4 conv-feature chunks) are loaded into VGPRs once and reused for all
//     32 d-tiles; B fragments stream from LDS one chunk ahead of the WMMA
//     that consumes them so DS latency overlaps the matrix pipe.
// ---------------------------------------------------------------------------
__global__ __launch_bounds__(THREADS1) void k1_scores(
    const float* __restrict__ enc,   const float* __restrict__ ali_prev,
    const float* __restrict__ Wkey,  const float* __restrict__ Fw,
    const float* __restrict__ attw,  const float* __restrict__ ww,
    const float* __restrict__ dec_ws, float* __restrict__ e_ws)
{
    extern __shared__ unsigned char smem[];
    unsigned short* lds_enc = (unsigned short*)(smem + OFF_ENC);
    unsigned short* lds_f   = (unsigned short*)(smem + OFF_F);
    unsigned short* lds_wk  = (unsigned short*)(smem + OFF_WK);
    unsigned short* lds_wa  = (unsigned short*)(smem + OFF_WA);
    float*          lds_fw  = (float*)(smem + OFF_FW);
    float*          lds_ali = (float*)(smem + OFF_ALI);

    const int tid  = threadIdx.x;
    const int wid  = tid >> 5;
    const int lane = tid & 31;

    int b = blockIdx.x;
    const int blk = b % NBLK; b /= NBLK;
    const int h   = b % H_;
    const int n   = b / H_;
    const int t0  = blk * TB;

    // ---- stage encoder tile fp32 -> bf16 (zero-fill rows past T) ----------
    for (int idx = tid; idx < TB * ENC_; idx += THREADS1) {
        const int r = idx >> 9, c = idx & (ENC_ - 1);
        const int t = t0 + r;
        const float v = (t < T_) ? enc[((size_t)n * T_ + t) * ENC_ + c] : 0.f;
        lds_enc[r * ENC_P + c] = f2bf(v);
    }
    // ---- stage conv weights + padded alignment window ---------------------
    for (int idx = tid; idx < C_ * K_; idx += THREADS1) {
        const int c = idx / K_, k = idx % K_;
        lds_fw[c * 12 + k] = Fw[(size_t)(h * C_ + c) * K_ + k];
    }
    for (int idx = tid; idx < TB + K_ - 1; idx += THREADS1) {
        const int t = t0 - PAD_ + idx;
        lds_ali[idx] = (t >= 0 && t < T_)
                     ? ali_prev[((size_t)n * H_ + h) * T_ + t] : 0.f;
    }
    __syncthreads();

    // ---- location features: f[t][c] = sum_k Fw[c,k] * ali[t-PAD+k] --------
    for (int idx = tid; idx < TB * C_; idx += THREADS1) {
        const int c = idx & (C_ - 1), r = idx >> 7;
        float a = 0.f;
        #pragma unroll
        for (int k = 0; k < K_; ++k)
            a += lds_fw[c * 12 + k] * lds_ali[r + k];
        lds_f[r * C_P + c] = f2bf(a);
    }
    __syncthreads();

    // ---- WMMA fragment addressing (wave32 layouts per CDNA5 ISA 7.12.2) ---
    const int mrow  = lane & 15;     // A row / B column owned by this lane
    const int khalf = lane >> 4;     // selects K sub-block of the fragment
    const int tw    = wid * 16;      // this wave's t-row base inside the tile

    // ---- hoist all A fragments into registers (reused for 32 d-tiles) ----
    Frag32 A[KC_ALL];
    {
        const unsigned short* pe = &lds_enc[(tw + mrow) * ENC_P + 8 * khalf];
        #pragma unroll
        for (int kc = 0; kc < KC_ENC; ++kc)
            load_a(pe + kc * 32, A[kc]);
        const unsigned short* pf = &lds_f[(tw + mrow) * C_P + 8 * khalf];
        #pragma unroll
        for (int cc = 0; cc < KC_F; ++cc)
            load_a(pf + cc * 32, A[KC_ENC + cc]);
    }

    float e_acc[8];
    #pragma unroll
    for (int r = 0; r < 8; ++r) e_acc[r] = 0.f;

    const unsigned short* pbk = &lds_wk[mrow * ENC_P + 8 * khalf];
    const unsigned short* pba = &lds_wa[mrow * C_P + 8 * khalf];

    for (int dt = 0; dt < D_ / 16; ++dt) {
        __syncthreads();   // previous d-tile's fragment reads are done
        // stage 16 columns of W_key for this (h, d-tile), K-major, as bf16
        for (int idx = tid; idx < 16 * ENC_; idx += THREADS1) {
            const int j = idx & 15, k = idx >> 4;
            lds_wk[j * ENC_P + k] =
                f2bf(Wkey[(size_t)k * (H_ * D_) + h * D_ + dt * 16 + j]);
        }
        // stage att_w[h, d-tile, :] (C-major per column) as bf16
        for (int idx = tid; idx < 16 * C_; idx += THREADS1) {
            const int j = idx >> 7, c = idx & (C_ - 1);
            lds_wa[j * C_P + c] =
                f2bf(attw[((size_t)h * D_ + dt * 16 + j) * C_ + c]);
        }
        // hint the next slice of W_key toward L2 while we compute
        if (dt + 1 < D_ / 16 && tid < 16)
            __builtin_prefetch(
                &Wkey[(size_t)(tid * 32) * (H_ * D_) + h * D_ + (dt + 1) * 16],
                0, 1);
        __syncthreads();

        // ---- 20-chunk contraction, B fragments prefetched one chunk ahead
        v8f acc = {};
        Frag32 bcur, bnxt;
        load_b(pbk, bcur);
        #pragma unroll
        for (int kc = 0; kc < KC_ENC; ++kc) {
            if (kc + 1 < KC_ENC) load_b(pbk + (kc + 1) * 32, bnxt);
            else                 load_b(pba, bnxt);
            acc = __builtin_amdgcn_wmma_f32_16x16x32_bf16(
                false, A[kc].v, false, bcur.v, (short)0, acc, false, false);
            bcur = bnxt;
        }
        #pragma unroll
        for (int cc = 0; cc < KC_F; ++cc) {
            if (cc + 1 < KC_F) load_b(pba + (cc + 1) * 32, bnxt);
            acc = __builtin_amdgcn_wmma_f32_16x16x32_bf16(
                false, A[KC_ENC + cc].v, false, bcur.v, (short)0, acc,
                false, false);
            bcur = bnxt;
        }

        // epilogue in C/D layout: lane holds column d = dt*16+mrow,
        // rows t = {0..7} + 8*khalf across the 8 accumulator VGPRs.
        const int   dcol = dt * 16 + mrow;
        const float decv = dec_ws[((size_t)n * H_ + h) * D_ + dcol];
        const float wv   = ww[h * D_ + dcol];
        #pragma unroll
        for (int r = 0; r < 8; ++r)
            e_acc[r] += wv * tanhf(acc[r] + decv);
    }

    // reduce over the 16 d-columns (lanes differing in bits 0..3 share a row)
    #pragma unroll
    for (int r = 0; r < 8; ++r) {
        float v = e_acc[r];
        v += __shfl_xor(v, 1, 32);
        v += __shfl_xor(v, 2, 32);
        v += __shfl_xor(v, 4, 32);
        v += __shfl_xor(v, 8, 32);
        e_acc[r] = v;
    }
    if (mrow == 0) {
        #pragma unroll
        for (int r = 0; r < 8; ++r) {
            const int t = t0 + tw + khalf * 8 + r;
            if (t < T_)
                e_ws[((size_t)n * H_ + h) * T_ + t] = e_acc[r];
        }
    }
}

// ---------------------------------------------------------------------------
// k2: masked softmax over T  ->  ali (output #1), then
//     we[n,h,e] = sum_t ali[n,h,t] * enc_pad[n,t,e]
// ---------------------------------------------------------------------------
__global__ __launch_bounds__(256) void k2_softmax_wenc(
    const float* __restrict__ e_ws, const int* __restrict__ enc_len,
    const float* __restrict__ enc,  float* __restrict__ ali_out,
    float* __restrict__ we_ws)
{
    __shared__ float s_p[T_];
    __shared__ float s_red[8];

    const int nh   = blockIdx.x;           // n*H + h
    const int n    = nh / H_;
    const int tid  = threadIdx.x;
    const int lane = tid & 31;
    const int wid  = tid >> 5;
    const int len  = enc_len[n];
    const float* erow = e_ws + (size_t)nh * T_;

    float mx = -__builtin_inff();
    for (int t = tid; t < T_; t += 256) {
        const float v = (t < len) ? erow[t] : -__builtin_inff();
        s_p[t] = v;
        mx = fmaxf(mx, v);
    }
    #pragma unroll
    for (int m = 16; m; m >>= 1) mx = fmaxf(mx, __shfl_xor(mx, m, 32));
    if (lane == 0) s_red[wid] = mx;
    __syncthreads();
    float gmx = s_red[0];
    #pragma unroll
    for (int i = 1; i < 8; ++i) gmx = fmaxf(gmx, s_red[i]);

    float sum = 0.f;
    for (int t = tid; t < T_; t += 256) {
        const float p = expf(s_p[t] - gmx);   // exp(-inf) == 0 handles mask
        s_p[t] = p;
        sum += p;
    }
    #pragma unroll
    for (int m = 16; m; m >>= 1) sum += __shfl_xor(sum, m, 32);
    __syncthreads();
    if (lane == 0) s_red[wid] = sum;
    __syncthreads();
    float tot = 0.f;
    #pragma unroll
    for (int i = 0; i < 8; ++i) tot += s_red[i];
    const float inv = 1.f / tot;

    for (int t = tid; t < T_; t += 256) {
        const float a = s_p[t] * inv;
        s_p[t] = a;
        ali_out[(size_t)nh * T_ + t] = a;
    }
    __syncthreads();

    // weighted encoder sum; t >= len contributes exactly zero, so skip it.
    const float* encn = enc + (size_t)n * T_ * ENC_;
    float a0 = 0.f, a1 = 0.f;
    for (int t = 0; t < len; ++t) {
        const float w = s_p[t];
        a0 += w * encn[(size_t)t * ENC_ + tid];
        a1 += w * encn[(size_t)t * ENC_ + tid + 256];
    }
    we_ws[(size_t)nh * ENC_ + tid]       = a0;
    we_ws[(size_t)nh * ENC_ + tid + 256] = a1;
}

// ---------------------------------------------------------------------------
// k3a: ctx_head[n, h*D+d] = we[n,h,:] @ W_enc[:, h*D+d] + b_enc  (sum ali == 1)
// ---------------------------------------------------------------------------
__global__ __launch_bounds__(256) void k3a_ctx_head(
    const float* __restrict__ we_ws, const float* __restrict__ Wenc,
    const float* __restrict__ benc,  float* __restrict__ ctx_head)
{
    const int gid = blockIdx.x * 256 + threadIdx.x;     // N*H*D
    const int n   = gid >> 11;
    const int col = gid & 2047;
    const int h   = col >> 9;
    const float* w = we_ws + ((size_t)n * H_ + h) * ENC_;
    float acc = benc[col];
    #pragma unroll 8
    for (int e = 0; e < ENC_; ++e)
        acc += w[e] * Wenc[(size_t)e * (H_ * D_) + col];
    ctx_head[gid] = acc;
}

// ---------------------------------------------------------------------------
// k3b: ctx[n, o] = ctx_head[n,:] @ W_ctx[:, o] + b_ctx   (output #2)
// ---------------------------------------------------------------------------
__global__ __launch_bounds__(256) void k3b_ctx(
    const float* __restrict__ ctx_head, const float* __restrict__ Wctx,
    const float* __restrict__ bctx,     float* __restrict__ ctx_out)
{
    const int gid = blockIdx.x * 256 + threadIdx.x;     // N*ENC
    const int n   = gid >> 9;
    const int o   = gid & 511;
    const float* c = ctx_head + (size_t)n * (H_ * D_);
    float acc = bctx[o];
    #pragma unroll 8
    for (int j = 0; j < H_ * D_; ++j)
        acc += c[j] * Wctx[(size_t)j * ENC_ + o];
    ctx_out[gid] = acc;
}

// ---------------------------------------------------------------------------
extern "C" void kernel_launch(void* const* d_in, const int* in_sizes, int n_in,
                              void* d_out, int out_size, void* d_ws,
                              size_t ws_size, hipStream_t stream)
{
    (void)in_sizes; (void)n_in; (void)out_size; (void)ws_size;

    const float* enc      = (const float*)d_in[0];
    const int*   enc_len  = (const int*)  d_in[1];
    const float* dec_prev = (const float*)d_in[2];
    const float* ali_prev = (const float*)d_in[3];
    const float* W_enc    = (const float*)d_in[4];
    const float* b_enc    = (const float*)d_in[5];
    const float* W_key    = (const float*)d_in[6];
    const float* W_dec    = (const float*)d_in[7];
    const float* F_w      = (const float*)d_in[8];
    const float* att_w    = (const float*)d_in[9];
    const float* w_w      = (const float*)d_in[10];
    const float* W_ctx    = (const float*)d_in[11];
    const float* b_ctx    = (const float*)d_in[12];

    float* out     = (float*)d_out;
    float* ali_out = out;                    // N*H*T
    float* ctx_out = out + (size_t)N_ * H_ * T_;

    float* ws       = (float*)d_ws;
    float* e_ws     = ws;                                  // N*H*T
    float* dec_ws   = e_ws   + (size_t)N_ * H_ * T_;       // N*H*D
    float* we_ws    = dec_ws + (size_t)N_ * H_ * D_;       // N*H*ENC
    float* ctx_head = we_ws  + (size_t)N_ * H_ * ENC_;     // N*H*D

    k0_dec_proj<<<(N_ * H_ * D_) / 256, 256, 0, stream>>>(
        dec_prev, W_dec, dec_ws);

    k1_scores<<<N_ * H_ * NBLK, THREADS1, SMEM1, stream>>>(
        enc, ali_prev, W_key, F_w, att_w, w_w, dec_ws, e_ws);

    k2_softmax_wenc<<<N_ * H_, 256, 0, stream>>>(
        e_ws, enc_len, enc, ali_out, we_ws);

    k3a_ctx_head<<<(N_ * H_ * D_) / 256, 256, 0, stream>>>(
        we_ws, W_enc, b_enc, ctx_head);

    k3b_ctx<<<(N_ * ENC_) / 256, 256, 0, stream>>>(
        ctx_head, W_ctx, b_ctx, ctx_out);
}
